// CodeExpressionContextMixer_2705829396959
// MI455X (gfx1250) — compile-verified, matrix operand order checked by reading, not run on hardware
//
#include <hip/hip_runtime.h>

typedef __attribute__((ext_vector_type(16))) _Float16 v16h;
typedef __attribute__((ext_vector_type(8)))  _Float16 v8h;
typedef __attribute__((ext_vector_type(4)))  _Float16 v4h;
typedef __attribute__((ext_vector_type(8)))  float    v8f;

#define AST_DIM 256
#define MBLK    64
// padded LDS row strides (in f16 elements) to avoid bank conflicts
#define S1 (256 + 8)   // ctx rows
#define S2 (512 + 8)   // [prev ; u] rows

#define LOG2E 1.4426950408889634f

// branch-free device transcendentals (native v_exp_f32 / v_rcp_f32 / v_tanh_f32)
__device__ __forceinline__ float fast_sigmoid(float x) {
    return __builtin_amdgcn_rcpf(1.0f + __builtin_amdgcn_exp2f(-x * LOG2E));
}
__device__ __forceinline__ float fast_tanh(float x) {
#if __has_builtin(__builtin_amdgcn_tanhf)
    return __builtin_amdgcn_tanhf(x);
#else
    // tanh(x) = 1 - 2/(exp(2x)+1); saturates correctly at +/-inf
    return 1.0f - 2.0f * __builtin_amdgcn_rcpf(
        __builtin_amdgcn_exp2f(2.0f * LOG2E * x) + 1.0f);
#endif
}

// ---------------------------------------------------------------------------
// Kernel 0a: zero the per-row "is mapped" flag array (re-done every call).
// ---------------------------------------------------------------------------
__global__ void zero_flags(unsigned int* __restrict__ flags32, int n32) {
    int i = blockIdx.x * blockDim.x + threadIdx.x;
    if (i < n32) flags32[i] = 0u;
}

// ---------------------------------------------------------------------------
// Kernel 0b: mark rows that will be overwritten by the scatter.
// ---------------------------------------------------------------------------
__global__ void mark_mapped(const int* __restrict__ mkey,
                            unsigned char* __restrict__ flags, int M) {
    int i = blockIdx.x * blockDim.x + threadIdx.x;
    if (i < M) flags[mkey[i]] = 1;
}

// ---------------------------------------------------------------------------
// Kernel 1: copy out = prev, but ONLY for rows not overwritten later.
// Mapped rows (80%) are skipped: their bytes are produced by the mixer's
// scatter, and their source data is read by the mixer's gather anyway.
// One float4 per thread-iteration; row = index>>6, so a wave spans <=2 rows
// and the skip branch is near-uniform.
// ---------------------------------------------------------------------------
__global__ void copy_unmapped(const float4* __restrict__ src,
                              float4* __restrict__ dst,
                              const unsigned char* __restrict__ flags, long n4) {
    long i = (long)blockIdx.x * blockDim.x + threadIdx.x;
    long stride = (long)gridDim.x * blockDim.x;
    for (; i < n4; i += stride) {
        long r = i >> 6;                  // 64 float4 per 256-float row
        if (!flags[r]) dst[i] = src[i];
    }
}

// ---------------------------------------------------------------------------
// Kernel 2: convert W_update [256(K)x256(N)] and W_gate [512(K)x256(N)] (f32,
// row-major) into f16 packed in per-lane WMMA B-operand order:
//   pack index p = (nt*KS + ks)*32 + lane ; 16 f16 per pack
//   elements 0..7  = B[k0+8g+j,   n]   (n = nt*16 + lane%16, g = lane/16)
//   elements 8..15 = B[k0+16+8g+j, n]  (k0 = ks*32)
// so a wave's B fetch for one (n-tile, k-step) is 1KB fully contiguous.
// ---------------------------------------------------------------------------
__global__ void pack_weights(const float* __restrict__ Wu,
                             const float* __restrict__ Wg,
                             _Float16* __restrict__ WuP,
                             _Float16* __restrict__ WgP) {
    int t = blockIdx.x * blockDim.x + threadIdx.x;
    if (t < 4096) {                       // Wu: 16 n-tiles * 8 k-steps * 32 lanes
        int nt = t >> 8, ks = (t >> 5) & 7, lane = t & 31;
        int lm = lane & 15, g = lane >> 4;
        int n = nt * 16 + lm, k0 = ks * 32;
        _Float16* dst = WuP + (size_t)t * 16;
#pragma unroll
        for (int j = 0; j < 8; ++j) {
            dst[j]     = (_Float16)Wu[(size_t)(k0 + 8 * g + j) * 256 + n];
            dst[8 + j] = (_Float16)Wu[(size_t)(k0 + 16 + 8 * g + j) * 256 + n];
        }
    } else if (t < 4096 + 8192) {         // Wg: 16 n-tiles * 16 k-steps * 32 lanes
        int p = t - 4096;
        int nt = p >> 9, ks = (p >> 5) & 15, lane = p & 31;
        int lm = lane & 15, g = lane >> 4;
        int n = nt * 16 + lm, k0 = ks * 32;
        _Float16* dst = WgP + (size_t)p * 16;
#pragma unroll
        for (int j = 0; j < 8; ++j) {
            dst[j]     = (_Float16)Wg[(size_t)(k0 + 8 * g + j) * 256 + n];
            dst[8 + j] = (_Float16)Wg[(size_t)(k0 + 16 + 8 * g + j) * 256 + n];
        }
    }
}

// ---------------------------------------------------------------------------
// Kernel 3: per 64-row M-block:
//   gather ctx/prev -> LDS (f16); u = tanh(ctx@Wu + bu) via WMMA -> LDS;
//   z = sigmoid([prev;u]@Wg + bg) via WMMA (K=512); blend; scatter to out.
// 8 wave32s; each wave owns n-tiles {w, w+8}; B packs reused over 4 m-tiles.
// launch_bounds(256,4): keep <=256 VGPRs (no vgpr_msb), 4 waves/SIMD.
// ---------------------------------------------------------------------------
__global__ __launch_bounds__(256, 4)
void mixer_kernel(const float* __restrict__ prev, const float* __restrict__ cfg,
                  const int* __restrict__ mkey, const int* __restrict__ mval,
                  const _Float16* __restrict__ WuP, const float* __restrict__ bu,
                  const _Float16* __restrict__ WgP, const float* __restrict__ bg,
                  float* __restrict__ out) {
    __shared__ _Float16 sA2[MBLK * S2];   // [m][0:256]=prev(f16), [256:512]=u(f16)
    __shared__ _Float16 sA1[MBLK * S1];   // [m][0:256]=ctx(f16)
    __shared__ int      sKey[MBLK];

    const int tid  = threadIdx.x;
    const int base = blockIdx.x * MBLK;

    // ---- gather: 4 threads per row, 64 cols each ----
    {
        int r = tid >> 2, q = tid & 3;
        int k = mkey[base + r];
        int v = mval[base + r];
        if (q == 0) sKey[r] = k;
        const float4* cs = (const float4*)(cfg  + (size_t)v * AST_DIM) + q * 16;
        const float4* ps = (const float4*)(prev + (size_t)k * AST_DIM) + q * 16;
        _Float16* a1 = sA1 + r * S1 + q * 64;
        _Float16* a2 = sA2 + r * S2 + q * 64;
#pragma unroll 4
        for (int i = 0; i < 16; ++i) {
            float4 c4 = cs[i];
            v4h hc = {(_Float16)c4.x, (_Float16)c4.y, (_Float16)c4.z, (_Float16)c4.w};
            *(v4h*)(a1 + i * 4) = hc;
            float4 p4 = ps[i];
            v4h hp = {(_Float16)p4.x, (_Float16)p4.y, (_Float16)p4.z, (_Float16)p4.w};
            *(v4h*)(a2 + i * 4) = hp;
        }
    }
    __syncthreads();

    const int w = tid >> 5, lane = tid & 31;
    const int lm = lane & 15, g = lane >> 4;

    // ---- GEMM1: u = tanh(ctx @ Wu + bu), K=256 ----
    for (int half = 0; half < 2; ++half) {
        int nt = w + 8 * half;
        int nrow = nt * 16 + lm;
        v8f acc[4] = {};
        const v8h* bp = (const v8h*)WuP + ((size_t)nt * 8) * 64 + lane * 2;
#pragma unroll 2
        for (int ks = 0; ks < 8; ++ks) {
            v8h b0 = bp[0], b1 = bp[1];
            bp += 64;
            v16h B = __builtin_shufflevector(b0, b1, 0, 1, 2, 3, 4, 5, 6, 7,
                                             8, 9, 10, 11, 12, 13, 14, 15);
            int k0 = ks * 32;
#pragma unroll
            for (int mt = 0; mt < 4; ++mt) {
                const _Float16* ap = sA1 + (mt * 16 + lm) * S1 + k0 + 8 * g;
                v8h a0 = *(const v8h*)ap;
                v8h a1 = *(const v8h*)(ap + 16);
                v16h A = __builtin_shufflevector(a0, a1, 0, 1, 2, 3, 4, 5, 6, 7,
                                                 8, 9, 10, 11, 12, 13, 14, 15);
                acc[mt] = __builtin_amdgcn_wmma_f32_16x16x32_f16(
                    false, A, false, B, (short)0, acc[mt], false, false);
            }
        }
        float bub = bu[nrow];
#pragma unroll
        for (int mt = 0; mt < 4; ++mt)
#pragma unroll
            for (int e = 0; e < 8; ++e) {
                int m = mt * 16 + e + 8 * g;
                float uval = fast_tanh(acc[mt][e] + bub);
                sA2[m * S2 + 256 + nrow] = (_Float16)uval;
            }
    }
    __syncthreads();

    // ---- GEMM2: z = sigmoid([prev;u] @ Wg + bg), K=512; blend + scatter ----
    for (int half = 0; half < 2; ++half) {
        int nt = w + 8 * half;
        int nrow = nt * 16 + lm;
        v8f acc[4] = {};
        const v8h* bp = (const v8h*)WgP + ((size_t)nt * 16) * 64 + lane * 2;
#pragma unroll 2
        for (int ks = 0; ks < 16; ++ks) {
            v8h b0 = bp[0], b1 = bp[1];
            bp += 64;
            v16h B = __builtin_shufflevector(b0, b1, 0, 1, 2, 3, 4, 5, 6, 7,
                                             8, 9, 10, 11, 12, 13, 14, 15);
            int k0 = ks * 32;
#pragma unroll
            for (int mt = 0; mt < 4; ++mt) {
                const _Float16* ap = sA2 + (mt * 16 + lm) * S2 + k0 + 8 * g;
                v8h a0 = *(const v8h*)ap;
                v8h a1 = *(const v8h*)(ap + 16);
                v16h A = __builtin_shufflevector(a0, a1, 0, 1, 2, 3, 4, 5, 6, 7,
                                                 8, 9, 10, 11, 12, 13, 14, 15);
                acc[mt] = __builtin_amdgcn_wmma_f32_16x16x32_f16(
                    false, A, false, B, (short)0, acc[mt], false, false);
            }
        }
        float bgb = bg[nrow];
#pragma unroll
        for (int mt = 0; mt < 4; ++mt)
#pragma unroll
            for (int e = 0; e < 8; ++e) {
                int m = mt * 16 + e + 8 * g;
                float z = fast_sigmoid(acc[mt][e] + bgb);
                float pv = (float)sA2[m * S2 + nrow];
                float uv = (float)sA2[m * S2 + 256 + nrow];
                out[(size_t)sKey[m] * AST_DIM + nrow] = z * pv + (1.0f - z) * uv;
            }
    }
}

// ---------------------------------------------------------------------------
extern "C" void kernel_launch(void* const* d_in, const int* in_sizes, int n_in,
                              void* d_out, int out_size, void* d_ws, size_t ws_size,
                              hipStream_t stream) {
    const float* prev = (const float*)d_in[0];
    const float* cfg  = (const float*)d_in[1];
    const int*   mkey = (const int*)d_in[2];
    const int*   mval = (const int*)d_in[3];
    const float* Wu   = (const float*)d_in[4];
    const float* bu   = (const float*)d_in[5];
    const float* Wg   = (const float*)d_in[6];
    const float* bg   = (const float*)d_in[7];
    float* out = (float*)d_out;

    const int M      = in_sizes[2];               // 400000 mapping entries
    const long nelem = (long)in_sizes[0];         // 500000*256
    const int  nrows = (int)(nelem / AST_DIM);    // 500000
    const long n4    = nelem / 4;                 // float4 count

    // workspace layout: [WuP 128KB][WgP 256KB][flags nrows bytes]
    _Float16* WuP = (_Float16*)d_ws;
    _Float16* WgP = WuP + 65536;
    unsigned char* flags = (unsigned char*)(WgP + 131072);

    int nflag32 = (nrows + 3) / 4;
    zero_flags<<<(nflag32 + 255) / 256, 256, 0, stream>>>((unsigned int*)flags, nflag32);
    mark_mapped<<<(M + 255) / 256, 256, 0, stream>>>(mkey, flags, M);

    copy_unmapped<<<8192, 256, 0, stream>>>((const float4*)prev, (float4*)out,
                                            flags, n4);

    pack_weights<<<(4096 + 8192 + 255) / 256, 256, 0, stream>>>(Wu, Wg, WuP, WgP);

    int nblocks = M / MBLK;                       // 6250 (M divisible by 64)
    mixer_kernel<<<nblocks, 256, 0, stream>>>(prev, cfg, mkey, mval,
                                              WuP, bu, WgP, bg, out);
}
